// SpatialLocalityAttention_36670430773467
// MI455X (gfx1250) — matrix operand order repaired and checked by reading, not compiled, not run on hardware
//
#include <hip/hip_runtime.h>
#include <hip/hip_bf16.h>
#include <cstdint>

#define S_LEN 2048
#define DM    1024
#define HEADS 16
#define DH    64

typedef __attribute__((ext_vector_type(16))) _Float16 v16h;
typedef __attribute__((ext_vector_type(8)))  _Float16 v8h;
typedef __attribute__((ext_vector_type(4)))  _Float16 v4h;
typedef __attribute__((ext_vector_type(8)))  float    v8f;

__device__ __forceinline__ v8f wmma_f16(v16h a, v16h b, v8f c) {
  // D = A(16x32,f16) x B(32x16,f16) + C(16x16,f32)
  return __builtin_amdgcn_wmma_f32_16x16x32_f16(false, a, false, b, (short)0, c,
                                                false, false);
}

// ---- CDNA5 async global->LDS copy (ASYNCcnt tracked, bypasses VGPRs) ------
__device__ __forceinline__ unsigned lds_off_of(const void* p) {
  // addrspace(3) pointer value == LDS byte offset
  return (unsigned)(uintptr_t)p;
}
__device__ __forceinline__ void async_copy16(unsigned lds_off, const void* gptr) {
  asm volatile("global_load_async_to_lds_b128 %0, %1, off"
               :: "v"(lds_off), "v"((unsigned long long)(uintptr_t)gptr)
               : "memory");
}
__device__ __forceinline__ void async_wait0() {
  asm volatile("s_wait_asynccnt 0x0" ::: "memory");
}

// A fragment (16x32 f16), source row-major; lane L holds row M=L&15,
// element i -> K = i + 16*(L>=16)
__device__ __forceinline__ v16h load_a_frag(const _Float16* __restrict__ row_ptr) {
  const int lane = threadIdx.x & 31;
  const int hi8  = (lane >> 4) << 3;
  v16h a;
#pragma unroll
  for (int i = 0; i < 16; ++i) {
    int k = i + 2 * hi8;
    a[i] = row_ptr[k];
  }
  return a;
}

// B fragment (32x16 f16) from source stored [N][K];
// lane L holds N=L&15, element i -> K = i + 16*(L>=16)
__device__ __forceinline__ v16h load_b_frag_nk(const _Float16* __restrict__ base,
                                               int row_stride, int nbase, int kbase) {
  const int lane = threadIdx.x & 31;
  const int koff = (lane >= 16) ? 16 : 0;
  const _Float16* p = base + (nbase + (lane & 15)) * row_stride + kbase + koff;
  v16h b;
#pragma unroll
  for (int i = 0; i < 16; ++i) b[i] = p[i];
  return b;
}

// ---------------------------------------------------------------------------
// One-time f32 -> f16 conversion (x and the four weight matrices)
// ---------------------------------------------------------------------------
__global__ __launch_bounds__(256) void cvt_f32_to_f16(const float* __restrict__ in,
                                                      _Float16* __restrict__ out,
                                                      int n4) {
  int i = blockIdx.x * 256 + threadIdx.x;
  if (i < n4) {
    float4 v = ((const float4*)in)[i];
    v4h h;
    h[0] = (_Float16)v.x; h[1] = (_Float16)v.y;
    h[2] = (_Float16)v.z; h[3] = (_Float16)v.w;
    ((v4h*)out)[i] = h;
  }
}

// ---------------------------------------------------------------------------
// Projection GEMM (all-f16, double-buffered async-to-LDS pipeline):
// Out[b,h,s,dh] (f16) = X[b,s,:] . W[e,:] + bias[e]
// Block = 8 waves; tile M=256 (32 rows/wave), N=64 (one head), K-step 32.
// ---------------------------------------------------------------------------
__global__ __launch_bounds__(256) void proj_gemm_f16(
    const _Float16* __restrict__ X, const _Float16* __restrict__ W,
    const float* __restrict__ bias, _Float16* __restrict__ Out) {
  __shared__ _Float16 Xs[2][256 * 40];   // 32 data halves + pad per row
  __shared__ _Float16 Ws[2][64 * 40];
  const int tid   = threadIdx.x;
  const int lane  = tid & 31;
  const int wave  = tid >> 5;
  const int m0    = blockIdx.x * 256;
  const int h     = blockIdx.y;
  const int nbase = h * 64;

  auto stage = [&](int buf, int kk) {
    // X tile: 256 rows x 32 halves = 1024 x 16B chunks; 4 per thread
#pragma unroll
    for (int j = 0; j < 4; ++j) {
      int idx = tid + j * 256;
      int r = idx >> 2, c = idx & 3;
      async_copy16(lds_off_of(&Xs[buf][r * 40 + c * 8]),
                   &X[(size_t)(m0 + r) * DM + kk + c * 8]);
    }
    // W tile: 64 x 32 halves = 256 chunks; 1 per thread
    int r = tid >> 2, c = tid & 3;
    async_copy16(lds_off_of(&Ws[buf][r * 40 + c * 8]),
                 &W[(size_t)(nbase + r) * DM + kk + c * 8]);
  };

  v8f acc[2][4] = {};
  stage(0, 0);
  for (int kk = 0, it = 0; kk < DM; kk += 32, ++it) {
    const int cur = it & 1;
    async_wait0();        // my tile-i copies complete (only ones outstanding)
    __syncthreads();      // publishes tile i to all waves
    if (kk + 32 < DM) stage(cur ^ 1, kk + 32);   // overlaps with compute below

    v16h a0 = load_a_frag(&Xs[cur][(wave * 32 +      (lane & 15)) * 40]);
    v16h a1 = load_a_frag(&Xs[cur][(wave * 32 + 16 + (lane & 15)) * 40]);
#pragma unroll
    for (int t = 0; t < 4; ++t) {
      v16h b = load_b_frag_nk(Ws[cur], 40, t * 16, 0);
      acc[0][t] = wmma_f16(a0, b, acc[0][t]);
      acc[1][t] = wmma_f16(a1, b, acc[1][t]);
    }
  }
  const int hi8 = (lane >> 4) << 3;
#pragma unroll
  for (int s2 = 0; s2 < 2; ++s2) {
#pragma unroll
    for (int t = 0; t < 4; ++t) {
      int dh = t * 16 + (lane & 15);
      float bb = bias[nbase + dh];
#pragma unroll
      for (int r = 0; r < 8; ++r) {
        int gm = m0 + wave * 32 + s2 * 16 + r + hi8;   // row in [B*S]
        int bi = gm >> 11, s = gm & (S_LEN - 1);
        Out[(size_t)((bi * HEADS + h) * S_LEN + s) * DH + dh] =
            (_Float16)(acc[s2][t][r] + bb);
      }
    }
  }
}

// ---------------------------------------------------------------------------
// Flash attention with spatial bias. Block = 8 waves, 128 q rows of one (b,h).
// Double-buffered staging: K async-to-LDS; V vector-loaded and stored
// transposed [dh][key] (PV B-frags become contiguous ds_load_b128); the
// next-tile V load is issued early and its LDS stores deferred past softmax.
// ---------------------------------------------------------------------------
__global__ __launch_bounds__(256) void attn_flash(
    const _Float16* __restrict__ Q, const _Float16* __restrict__ K,
    const _Float16* __restrict__ V, const float* __restrict__ coords,
    const float* __restrict__ logbw, _Float16* __restrict__ Aout) {
  __shared__ _Float16 Ks[2][32 * 72];     // [key][dh], padded
  __shared__ _Float16 Vt[2][64 * 36];     // [dh][key], padded (transposed V)
  __shared__ float    Kc[2][64];
  __shared__ _Float16 Ps[8][16 * 34];     // per-wave P scratch (C->A relayout)
  const int tid  = threadIdx.x;
  const int lane = tid & 31;
  const int wave = tid >> 5;
  const int bh = blockIdx.y;
  const int b = bh >> 4, h = bh & 15;
  const int q0 = blockIdx.x * 128 + wave * 16;
  const size_t head_off = (size_t)(b * HEADS + h) * S_LEN * DH;
  const _Float16* Qbh = Q + head_off;
  const _Float16* Kbh = K + head_off;
  const _Float16* Vbh = V + head_off;
  const int hi8 = (lane >> 4) << 3;

  v16h qa0 = load_a_frag(&Qbh[(size_t)(q0 + (lane & 15)) * DH]);
  v16h qa1 = load_a_frag(&Qbh[(size_t)(q0 + (lane & 15)) * DH + 32]);

  float qcx[8], qcy[8];
#pragma unroll
  for (int r = 0; r < 8; ++r) {
    int m = q0 + r + hi8;
    qcx[r] = coords[((size_t)b * S_LEN + m) * 2 + 0];
    qcy[r] = coords[((size_t)b * S_LEN + m) * 2 + 1];
  }
  float bw = __expf(logbw[h]);
  const float negInv = -0.5f / (bw * bw);
  const float scale  = 0.125f;          // 1/sqrt(64)

  v8f o[4] = {};
  float mi[8], li[8];
#pragma unroll
  for (int r = 0; r < 8; ++r) { mi[r] = -3.0e38f; li[r] = 0.0f; }

  const int kr = tid >> 3, ck = tid & 7;   // 32 rows x 8 chunks, 1 per thread

  // ---- prologue: stage key-tile 0 into buffer 0 ----
  async_copy16(lds_off_of(&Ks[0][kr * 72 + ck * 8]),
               &Kbh[(size_t)kr * DH + ck * 8]);
  {
    v8h vv = *(const v8h*)&Vbh[(size_t)kr * DH + ck * 8];
#pragma unroll
    for (int j = 0; j < 8; ++j) Vt[0][(ck * 8 + j) * 36 + kr] = vv[j];
  }
  if (tid < 64) Kc[0][tid] = coords[((size_t)b * S_LEN) * 2 + tid];

  for (int kb = 0, it = 0; kb < S_LEN; kb += 32, ++it) {
    const int cur = it & 1, nxt = cur ^ 1;
    async_wait0();
    __syncthreads();      // publishes Ks/Vt/Kc[cur]
    const bool more = (kb + 32 < S_LEN);
    v8h vvn;
    if (more) {
      // prefetch next tile: K via async engine, V into registers
      async_copy16(lds_off_of(&Ks[nxt][kr * 72 + ck * 8]),
                   &Kbh[(size_t)(kb + 32 + kr) * DH + ck * 8]);
      vvn = *(const v8h*)&Vbh[(size_t)(kb + 32 + kr) * DH + ck * 8];
      if (tid < 64) Kc[nxt][tid] = coords[((size_t)b * S_LEN + kb + 32) * 2 + tid];
    }

    // logits S = Q . K^T  (two 16x16 N-tiles, K-dim 64 = 2 x 32)
    v8f st[2];
#pragma unroll
    for (int nt = 0; nt < 2; ++nt) {
      v8f s0 = {};
      s0 = wmma_f16(qa0, load_b_frag_nk(Ks[cur], 72, nt * 16, 0),  s0);
      s0 = wmma_f16(qa1, load_b_frag_nk(Ks[cur], 72, nt * 16, 32), s0);
      st[nt] = s0;
    }

    // spatial bias + online softmax (C layout: M=r+hi8, N=lane&15)
    float p[2][8], rmax[8];
#pragma unroll
    for (int r = 0; r < 8; ++r) rmax[r] = -3.0e38f;
#pragma unroll
    for (int nt = 0; nt < 2; ++nt) {
      int nl = nt * 16 + (lane & 15);
      float kx = Kc[cur][nl * 2 + 0], ky = Kc[cur][nl * 2 + 1];
#pragma unroll
      for (int r = 0; r < 8; ++r) {
        float dx = qcx[r] - kx, dy = qcy[r] - ky;
        float lg = st[nt][r] * scale + (dx * dx + dy * dy) * negInv;
        p[nt][r] = lg;
        rmax[r] = fmaxf(rmax[r], lg);
      }
    }
#pragma unroll
    for (int r = 0; r < 8; ++r) {
      float v = rmax[r];
#pragma unroll
      for (int msk = 1; msk < 16; msk <<= 1)
        v = fmaxf(v, __shfl_xor(v, msk, 32));   // stays within 16-lane half
      float mnew  = fmaxf(mi[r], v);
      float alpha = __expf(mi[r] - mnew);
      mi[r] = mnew;
      float e0 = __expf(p[0][r] - mnew);
      float e1 = __expf(p[1][r] - mnew);
      p[0][r] = e0; p[1][r] = e1;
      float rs = e0 + e1;
#pragma unroll
      for (int msk = 1; msk < 16; msk <<= 1) rs += __shfl_xor(rs, msk, 32);
      li[r] = li[r] * alpha + rs;
#pragma unroll
      for (int t = 0; t < 4; ++t) o[t][r] *= alpha;
    }

    // deferred transposed store of next V tile (global latency now hidden)
    if (more) {
#pragma unroll
      for (int j = 0; j < 8; ++j) Vt[nxt][(ck * 8 + j) * 36 + kr] = vvn[j];
    }

    // P (C layout) -> A fragment via wave-private LDS (DS in-order per wave)
#pragma unroll
    for (int nt = 0; nt < 2; ++nt) {
      int nl = nt * 16 + (lane & 15);
#pragma unroll
      for (int r = 0; r < 8; ++r)
        Ps[wave][(r + hi8) * 34 + nl] = (_Float16)p[nt][r];
    }
    asm volatile("" ::: "memory");
    v16h pa = load_a_frag(&Ps[wave][(lane & 15) * 34]);

    // O += P(16x32) x V(32x64); B-frags from Vt are contiguous b128 loads
#pragma unroll
    for (int t = 0; t < 4; ++t)
      o[t] = wmma_f16(pa, load_b_frag_nk(Vt[cur], 36, t * 16, 0), o[t]);
  }

#pragma unroll
  for (int r = 0; r < 8; ++r) li[r] = 1.0f / li[r];
#pragma unroll
  for (int t = 0; t < 4; ++t) {
    int col = h * DH + t * 16 + (lane & 15);
#pragma unroll
    for (int r = 0; r < 8; ++r) {
      int srow = q0 + r + hi8;
      Aout[((size_t)b * S_LEN + srow) * DM + col] = (_Float16)(o[t][r] * li[r]);
    }
  }
}

// ---------------------------------------------------------------------------
// Output projection: d_out (f32) = A(f16)[B*S,1024] . Wo^T + bo
// Same double-buffered async pipeline, f32 output.
// ---------------------------------------------------------------------------
__global__ __launch_bounds__(256) void out_gemm_f32(
    const _Float16* __restrict__ A, const _Float16* __restrict__ W,
    const float* __restrict__ bias, float* __restrict__ Out) {
  __shared__ _Float16 As[2][256 * 40];
  __shared__ _Float16 Ws[2][64 * 40];
  const int tid   = threadIdx.x;
  const int lane  = tid & 31;
  const int wave  = tid >> 5;
  const int m0    = blockIdx.x * 256;
  const int nbase = blockIdx.y * 64;

  auto stage = [&](int buf, int kk) {
#pragma unroll
    for (int j = 0; j < 4; ++j) {
      int idx = tid + j * 256;
      int r = idx >> 2, c = idx & 3;
      async_copy16(lds_off_of(&As[buf][r * 40 + c * 8]),
                   &A[(size_t)(m0 + r) * DM + kk + c * 8]);
    }
    int r = tid >> 2, c = tid & 3;
    async_copy16(lds_off_of(&Ws[buf][r * 40 + c * 8]),
                 &W[(size_t)(nbase + r) * DM + kk + c * 8]);
  };

  v8f acc[2][4] = {};
  stage(0, 0);
  for (int kk = 0, it = 0; kk < DM; kk += 32, ++it) {
    const int cur = it & 1;
    async_wait0();
    __syncthreads();
    if (kk + 32 < DM) stage(cur ^ 1, kk + 32);

    v16h a0 = load_a_frag(&As[cur][(wave * 32 +      (lane & 15)) * 40]);
    v16h a1 = load_a_frag(&As[cur][(wave * 32 + 16 + (lane & 15)) * 40]);
#pragma unroll
    for (int t = 0; t < 4; ++t) {
      v16h b = load_b_frag_nk(Ws[cur], 40, t * 16, 0);
      acc[0][t] = wmma_f16(a0, b, acc[0][t]);
      acc[1][t] = wmma_f16(a1, b, acc[1][t]);
    }
  }
  const int hi8 = (lane >> 4) << 3;
#pragma unroll
  for (int s2 = 0; s2 < 2; ++s2) {
#pragma unroll
    for (int t = 0; t < 4; ++t) {
      int n = nbase + t * 16 + (lane & 15);
      float bb = bias[n];
#pragma unroll
      for (int r = 0; r < 8; ++r) {
        int gm = m0 + wave * 32 + s2 * 16 + r + hi8;
        Out[(size_t)gm * DM + n] = acc[s2][t][r] + bb;
      }
    }
  }
}

extern "C" void kernel_launch(void* const* d_in, const int* in_sizes, int n_in,
                              void* d_out, int out_size, void* d_ws, size_t ws_size,
                              hipStream_t stream) {
  const float* x      = (const float*)d_in[0];
  const float* coords = (const float*)d_in[1];
  const float* Wq = (const float*)d_in[2];  const float* bq = (const float*)d_in[3];
  const float* Wk = (const float*)d_in[4];  const float* bk = (const float*)d_in[5];
  const float* Wv = (const float*)d_in[6];  const float* bv = (const float*)d_in[7];
  const float* Wo = (const float*)d_in[8];  const float* bo = (const float*)d_in[9];
  const float* lbw = (const float*)d_in[10];

  const int B = in_sizes[0] / (S_LEN * DM);          // == 4
  const size_t elems  = (size_t)B * S_LEN * DM;      // 8,388,608
  const size_t welems = (size_t)DM * DM;             // 1,048,576

  _Float16* Xh  = (_Float16*)d_ws;       // x as f16      [B*S, D]
  _Float16* Qh  = Xh  + elems;           // [B,H,S,Dh]
  _Float16* Kh  = Qh  + elems;
  _Float16* Vh  = Kh  + elems;
  _Float16* Ah  = Vh  + elems;           // attn out f16  [B,S,D]
  _Float16* Wqh = Ah  + elems;           // weights as f16
  _Float16* Wkh = Wqh + welems;
  _Float16* Wvh = Wkh + welems;
  _Float16* Woh = Wvh + welems;

  dim3 blk(256);
  cvt_f32_to_f16<<<dim3((unsigned)(elems  / 4 / 256)), blk, 0, stream>>>(x,  Xh,  (int)(elems  / 4));
  cvt_f32_to_f16<<<dim3((unsigned)(welems / 4 / 256)), blk, 0, stream>>>(Wq, Wqh, (int)(welems / 4));
  cvt_f32_to_f16<<<dim3((unsigned)(welems / 4 / 256)), blk, 0, stream>>>(Wk, Wkh, (int)(welems / 4));
  cvt_f32_to_f16<<<dim3((unsigned)(welems / 4 / 256)), blk, 0, stream>>>(Wv, Wvh, (int)(welems / 4));
  cvt_f32_to_f16<<<dim3((unsigned)(welems / 4 / 256)), blk, 0, stream>>>(Wo, Woh, (int)(welems / 4));

  dim3 gproj((B * S_LEN) / 256, DM / 64);
  proj_gemm_f16<<<gproj, blk, 0, stream>>>(Xh, Wqh, bq, Qh);
  proj_gemm_f16<<<gproj, blk, 0, stream>>>(Xh, Wkh, bk, Kh);
  proj_gemm_f16<<<gproj, blk, 0, stream>>>(Xh, Wvh, bv, Vh);

  dim3 gattn(S_LEN / 128, B * HEADS);
  attn_flash<<<gattn, blk, 0, stream>>>(Qh, Kh, Vh, coords, lbw, Ah);

  out_gemm_f32<<<gproj, blk, 0, stream>>>(Ah, Woh, bo, (float*)d_out);
}
